// FERN_19241453486512
// MI455X (gfx1250) — compile-verified, matrix operand order checked by reading, not run on hardware
//
#include <hip/hip_runtime.h>
#include <cmath>

// ---------------------------------------------------------------------------
// Shapes
// ---------------------------------------------------------------------------
#define Bb 64
#define Cc 64
#define Ll 720
#define LlP 736              // Ll padded to multiple of 32 (K padding)
#define Dd 512
#define Pp 336
#define P2 168
#define Hh 512
#define Mm (Bb * Cc)         // 4096 flattened rows

// 256 threads = 8 wave32 waves. Block tile 256(M) x 16(N) of each output half.
// Each wave owns two 16-row A fragments -> 4 WMMA per K-step in the fused kernel.
#define BM   256
#define TPB  256

typedef __attribute__((ext_vector_type(16))) __bf16    v16bf;
typedef __attribute__((ext_vector_type(8)))  float     v8f;
typedef __attribute__((ext_vector_type(4)))  float     f32x4;
typedef __attribute__((ext_vector_type(4)))  unsigned  u32x4;

union FragAB { v16bf v; unsigned u[8]; u32x4 q[2]; };
union FragC  { v8f   v; float    f[8]; };

__device__ __forceinline__ unsigned short f2bf(float x) {
  unsigned u = __float_as_uint(x);
  u += 0x7FFFu + ((u >> 16) & 1u);          // round-to-nearest-even
  return (unsigned short)(u >> 16);
}
__device__ __forceinline__ unsigned pk2bf(float lo, float hi) {
  return (unsigned)f2bf(lo) | ((unsigned)f2bf(hi) << 16);
}

// All matrix operands live in global memory pre-packed in WMMA fragment order:
// per (tile, kt): 32 lanes x 8 dwords contiguous (1 KB). A fragment load is two
// 16B-aligned b128 loads straight into the WMMA source VGPRs.
__device__ __forceinline__ void load_frag8(FragAB& f, const unsigned* p) {
  f.q[0] = ((const u32x4*)p)[0];
  f.q[1] = ((const u32x4*)p)[1];
}

#define WMMA_BF16(A, B, C) \
  __builtin_amdgcn_wmma_f32_16x16x32_bf16(false, (A).v, false, (B).v, (short)0, (C).v, false, false)

// ---------------------------------------------------------------------------
// Fused affine GEMM:  v <- v * exp(tanh(cond @ W[:, :NH])) + cond @ W[:, NH:]
// Af: fragment-packed bf16 of cond [M,KP]; Wf: fragment-packed bf16 of W [K,2NH]
// ---------------------------------------------------------------------------
template<int KP, int NH, int VS>
__global__ __launch_bounds__(TPB) void affine_gemm(const unsigned* __restrict__ Af,
                                                   const unsigned* __restrict__ Wf,
                                                   const float* vin, float* vout) {
  constexpr int KTiles = KP / 32;
  const int lane = threadIdx.x & 31, wave = threadIdx.x >> 5;
  const int nt  = blockIdx.y;
  const int mt0 = blockIdx.x * (BM / 16) + wave * 2;

  FragC s0, t0, s1, t1;
#pragma unroll
  for (int r = 0; r < 8; ++r) { s0.f[r] = 0.f; t0.f[r] = 0.f; s1.f[r] = 0.f; t1.f[r] = 0.f; }

  const unsigned* pa0 = Af + (((size_t)mt0 * KTiles) * 32 + lane) * 8;
  const unsigned* pa1 = pa0 + (size_t)KTiles * 256;
  const unsigned* pbs = Wf + (((size_t)nt * KTiles) * 32 + lane) * 8;
  const unsigned* pbt = Wf + ((((size_t)nt + NH / 16) * KTiles) * 32 + lane) * 8;

  for (int kt = 0; kt < KTiles; ++kt) {
    FragAB a0, a1, bs, bt;
    load_frag8(a0, pa0 + (size_t)kt * 256);
    load_frag8(a1, pa1 + (size_t)kt * 256);
    load_frag8(bs, pbs + (size_t)kt * 256);
    load_frag8(bt, pbt + (size_t)kt * 256);
    if (kt + 1 < KTiles) {                 // warm caches for the next fragment block
      __builtin_prefetch(pbs + (size_t)(kt + 1) * 256, 0, 1);
      __builtin_prefetch(pbt + (size_t)(kt + 1) * 256, 0, 1);
    }
    s0.v = WMMA_BF16(a0, bs, s0);
    t0.v = WMMA_BF16(a0, bt, t0);
    s1.v = WMMA_BF16(a1, bs, s1);
    t1.v = WMMA_BF16(a1, bt, t1);
  }

  // C/D layout: VGPR r -> (m = r + 8*(lane>>4), n = lane&15)
  const int col = nt * 16 + (lane & 15);
  const int mb  = mt0 * 16 + ((lane >> 4) << 3);
#pragma unroll
  for (int r = 0; r < 8; ++r) {
    const size_t i0 = (size_t)(mb + r) * VS + col;
    const size_t i1 = (size_t)(mb + 16 + r) * VS + col;
    vout[i0] = vin[i0] * expf(tanhf(s0.f[r])) + t0.f[r];
    vout[i1] = vin[i1] * expf(tanhf(s1.f[r])) + t1.f[r];
  }
}

// ---------------------------------------------------------------------------
// GEMM + gelu(tanh approx): out[M,N] = gelu(cond @ W)
// ---------------------------------------------------------------------------
template<int KP, int N>
__global__ __launch_bounds__(TPB) void gemm_gelu(const unsigned* __restrict__ Af,
                                                 const unsigned* __restrict__ Wf,
                                                 float* __restrict__ out) {
  constexpr int KTiles = KP / 32;
  const int lane = threadIdx.x & 31, wave = threadIdx.x >> 5;
  const int nt  = blockIdx.y;
  const int mt0 = blockIdx.x * (BM / 16) + wave * 2;

  FragC c0, c1;
#pragma unroll
  for (int r = 0; r < 8; ++r) { c0.f[r] = 0.f; c1.f[r] = 0.f; }
  const unsigned* pa0 = Af + (((size_t)mt0 * KTiles) * 32 + lane) * 8;
  const unsigned* pa1 = pa0 + (size_t)KTiles * 256;
  const unsigned* pb  = Wf + (((size_t)nt * KTiles) * 32 + lane) * 8;

  for (int kt = 0; kt < KTiles; ++kt) {
    FragAB a0, a1, b;
    load_frag8(a0, pa0 + (size_t)kt * 256);
    load_frag8(a1, pa1 + (size_t)kt * 256);
    load_frag8(b,  pb  + (size_t)kt * 256);
    if (kt + 1 < KTiles) __builtin_prefetch(pb + (size_t)(kt + 1) * 256, 0, 1);
    c0.v = WMMA_BF16(a0, b, c0);
    c1.v = WMMA_BF16(a1, b, c1);
  }
  const int col = nt * 16 + (lane & 15);
  const int mb  = mt0 * 16 + ((lane >> 4) << 3);
#pragma unroll
  for (int r = 0; r < 8; ++r) {
    const float x0 = c0.f[r], x1 = c1.f[r];
    out[(size_t)(mb + r) * N + col] =
        0.5f * x0 * (1.f + tanhf(0.7978845608028654f * (x0 + 0.044715f * x0 * x0 * x0)));
    out[(size_t)(mb + 16 + r) * N + col] =
        0.5f * x1 * (1.f + tanhf(0.7978845608028654f * (x1 + 0.044715f * x1 * x1 * x1)));
  }
}

// ---------------------------------------------------------------------------
// Plain GEMM (N not necessarily multiple of 16): out[M,N] = cond @ W
// ---------------------------------------------------------------------------
template<int KP, int N>
__global__ __launch_bounds__(TPB) void gemm_plain(const unsigned* __restrict__ Af,
                                                  const unsigned* __restrict__ Wf,
                                                  float* __restrict__ out) {
  constexpr int KTiles = KP / 32;
  const int lane = threadIdx.x & 31, wave = threadIdx.x >> 5;
  const int nt  = blockIdx.y;
  const int mt0 = blockIdx.x * (BM / 16) + wave * 2;

  FragC c0, c1;
#pragma unroll
  for (int r = 0; r < 8; ++r) { c0.f[r] = 0.f; c1.f[r] = 0.f; }
  const unsigned* pa0 = Af + (((size_t)mt0 * KTiles) * 32 + lane) * 8;
  const unsigned* pa1 = pa0 + (size_t)KTiles * 256;
  const unsigned* pb  = Wf + (((size_t)nt * KTiles) * 32 + lane) * 8;

  for (int kt = 0; kt < KTiles; ++kt) {
    FragAB a0, a1, b;
    load_frag8(a0, pa0 + (size_t)kt * 256);
    load_frag8(a1, pa1 + (size_t)kt * 256);
    load_frag8(b,  pb  + (size_t)kt * 256);
    c0.v = WMMA_BF16(a0, b, c0);
    c1.v = WMMA_BF16(a1, b, c1);
  }
  const int col = nt * 16 + (lane & 15);
  const int mb  = mt0 * 16 + ((lane >> 4) << 3);
  if (col < N) {
#pragma unroll
    for (int r = 0; r < 8; ++r) {
      out[(size_t)(mb + r) * N + col]      = c0.f[r];
      out[(size_t)(mb + 16 + r) * N + col] = c1.f[r];
    }
  }
}

// ---------------------------------------------------------------------------
// Weight pre-pack: fp32 W[K,N] -> bf16 B-fragments
//   out[((nt*KTiles + kt)*32 + lane)*8 + j]   (zero-padded K and N tails)
// ---------------------------------------------------------------------------
__global__ void packW(const float* __restrict__ W, unsigned* __restrict__ out,
                      int K, int N, int KTiles, int NTiles) {
  const int g = blockIdx.x * blockDim.x + threadIdx.x;
  if (g >= NTiles * KTiles * 32) return;
  const int lane = g & 31;
  const int t = g >> 5;
  const int kt = t % KTiles;
  const int nt = t / KTiles;
  const int col = nt * 16 + (lane & 15);
  const int kbH = (lane >> 4) << 3;
  unsigned* dst = out + (size_t)g * 8;
#pragma unroll
  for (int j = 0; j < 8; ++j) {
    const int kk = ((j & 4) << 2) + kbH + ((j & 3) << 1);
    const int k = kt * 32 + kk;
    const bool cok = (col < N);
    const float v0 = (cok && k     < K) ? W[(size_t)k * N + col]       : 0.f;
    const float v1 = (cok && k + 1 < K) ? W[(size_t)(k + 1) * N + col] : 0.f;
    dst[j] = pk2bf(v0, v1);
  }
}

// ---------------------------------------------------------------------------
// Activation pre-pack: fp32 src[M,KP] (K already zero-padded) -> bf16 A-fragments
// Each thread = one lane of one (mt, kt) tile: 4x b128 loads, 2x b128 stores.
// ---------------------------------------------------------------------------
template<int KP>
__global__ __launch_bounds__(TPB) void packA(const float* __restrict__ src,
                                             unsigned* __restrict__ dst) {
  constexpr int KTiles = KP / 32;
  const int g = blockIdx.x * blockDim.x + threadIdx.x;
  if (g >= (Mm / 16) * KTiles * 32) return;
  const int lane = g & 31;
  const int t = g >> 5;
  const int kt = t % KTiles;
  const int mt = t / KTiles;
  const int row = mt * 16 + (lane & 15);
  const int kb4 = ((lane >> 4) << 3) >> 2;          // {0,2} in float4 units
  const f32x4* p = (const f32x4*)&src[(size_t)row * KP + kt * 32];
  const f32x4 v0 = p[kb4], v1 = p[kb4 + 1], v2 = p[4 + kb4], v3 = p[4 + kb4 + 1];
  u32x4 d0, d1;
  d0[0] = pk2bf(v0[0], v0[1]); d0[1] = pk2bf(v0[2], v0[3]);
  d0[2] = pk2bf(v1[0], v1[1]); d0[3] = pk2bf(v1[2], v1[3]);
  d1[0] = pk2bf(v2[0], v2[1]); d1[1] = pk2bf(v2[2], v2[3]);
  d1[2] = pk2bf(v3[0], v3[1]); d1[3] = pk2bf(v3[2], v3[3]);
  u32x4* o = (u32x4*)(dst + (size_t)g * 8);
  o[0] = d0;
  o[1] = d1;
}

// ---------------------------------------------------------------------------
// Prep kernels
// ---------------------------------------------------------------------------
__global__ void prep_x(const float* __restrict__ xb, float* __restrict__ X, int n) {
  const int idx = blockIdx.x * blockDim.x + threadIdx.x;
  if (idx >= n) return;
  const int l = idx % LlP;
  const int m = idx / LlP;
  const int b = m >> 6, c = m & 63;
  X[idx] = (l < Ll) ? xb[((size_t)b * Ll + l) * Cc + c] : 0.f;   // (B,L,C)->(M,LlP)
}

__global__ void scale01(const float* __restrict__ in, float* __restrict__ out, int n) {
  const int idx = blockIdx.x * blockDim.x + threadIdx.x;
  if (idx < n) out[idx] = 0.1f * in[idx];
}

// ---------------------------------------------------------------------------
// Final stage: theta/koopman pair math + transposed (B,P,C) store
// ---------------------------------------------------------------------------
__global__ void final_mix(const float* __restrict__ Y,   // [M,P]  pre_y
                          const float* __restrict__ R,   // [M,P2] z @ W_rot
                          const float* __restrict__ G,   // [M,2P] z @ W_koo
                          const float* __restrict__ a_y, const float* __restrict__ b_y,
                          float* __restrict__ out, int n) {
  const int idx = blockIdx.x * blockDim.x + threadIdx.x;
  if (idx >= n) return;
  const int j = idx % P2;
  const int m = idx / P2;
  const int b = m >> 6, c = m & 63;

  const float th = 3.14159265358979323846f * tanhf(R[(size_t)m * P2 + j]);
  const float co = cosf(th), si = sinf(th);
  const float a  = a_y[(size_t)c * P2 + j];
  const float bv = b_y[(size_t)c * P2 + j];
  const float inv_n = rsqrtf(a * a + bv * bv);
  const float ca = a * inv_n, cb = bv * inv_n;

  const float re = Y[(size_t)m * Pp + 2 * j];
  const float im = Y[(size_t)m * Pp + 2 * j + 1];
  const float r1 = co * re - si * im, i1 = si * re + co * im;        // rot
  const float r2 = ca * r1 - cb * i1, i2 = cb * r1 + ca * i1;        // koop
  const float s0 = expf(tanhf(G[(size_t)m * 672 + 2 * j]));
  const float s1 = expf(tanhf(G[(size_t)m * 672 + 2 * j + 1]));
  const float r3 = r2 * s0, i3 = i2 * s1;                            // * scale
  const float r4 = ca * r3 + cb * i3, i4 = ca * i3 - cb * r3;        // koop^-1
  const float r5 = co * r4 + si * i4, i5 = co * i4 - si * r4;        // rot^-1
  const float sh0 = G[(size_t)m * 672 + Pp + 2 * j];
  const float sh1 = G[(size_t)m * 672 + Pp + 2 * j + 1];
  out[((size_t)b * Pp + 2 * j) * Cc + c]     = r5 + sh0;
  out[((size_t)b * Pp + 2 * j + 1) * Cc + c] = i5 + sh1;
}

// ---------------------------------------------------------------------------
// Launcher
// ---------------------------------------------------------------------------
extern "C" void kernel_launch(void* const* d_in, const int* in_sizes, int n_in,
                              void* d_out, int out_size, void* d_ws, size_t ws_size,
                              hipStream_t stream) {
  const float* x_bsd   = (const float*)d_in[0];
  const float* z0      = (const float*)d_in[1];
  const float* y0      = (const float*)d_in[2];
  const float* W_h     = (const float*)d_in[3];
  const float* W_ssz   = (const float*)d_in[4];
  const float* W_xz_v2 = (const float*)d_in[5];
  const float* W_xz_v3 = (const float*)d_in[6];
  const float* W_xz_v4 = (const float*)d_in[7];
  const float* W_xz_v5 = (const float*)d_in[8];
  const float* W_zx_v0 = (const float*)d_in[9];
  const float* W_zx_v2 = (const float*)d_in[10];
  const float* W_zx_v3 = (const float*)d_in[11];
  // d_in[12] = W_zx_v5: produces a dead x update -> skipped
  const float* W_zy_v4 = (const float*)d_in[13];
  const float* W_rot   = (const float*)d_in[14];
  const float* W_koo   = (const float*)d_in[15];
  const float* a_y     = (const float*)d_in[16];
  const float* b_y     = (const float*)d_in[17];

  float* ws = (float*)d_ws;
  float* X = ws;                          // 4096 x 736
  float* Z = X + (size_t)Mm * LlP;        // 4096 x 512
  float* Y = Z + (size_t)Mm * Dd;         // 4096 x 336
  float* H = Y + (size_t)Mm * Pp;         // 4096 x 512
  float* R = H + (size_t)Mm * Hh;         // 4096 x 168
  float* G = R + (size_t)Mm * P2;         // 4096 x 672
  unsigned* Wp = (unsigned*)(G + (size_t)Mm * 672);

  // fragment-pack sizes (dwords) = NTiles * KTiles * 256
  const int KT5 = 16, KT7 = 23;           // K=512 -> 16 tiles, K=736 -> 23 tiles
  unsigned* WfH   = Wp;                                 // 32 x 23
  unsigned* Wfssz = WfH   + (size_t)32 * KT7 * 256;     // 64 x 16
  unsigned* Wfzx0 = Wfssz + (size_t)64 * KT5 * 256;     // 90 x 16
  unsigned* Wfxz2 = Wfzx0 + (size_t)90 * KT5 * 256;     // 64 x 23
  unsigned* Wfzx2 = Wfxz2 + (size_t)64 * KT7 * 256;     // 90 x 16
  unsigned* Wfxz3 = Wfzx2 + (size_t)90 * KT5 * 256;     // 64 x 23
  unsigned* Wfzx3 = Wfxz3 + (size_t)64 * KT7 * 256;     // 90 x 16
  unsigned* Wfxz4 = Wfzx3 + (size_t)90 * KT5 * 256;     // 64 x 23
  unsigned* Wfzy4 = Wfxz4 + (size_t)64 * KT7 * 256;     // 42 x 16
  unsigned* Wfxz5 = Wfzy4 + (size_t)42 * KT5 * 256;     // 64 x 23
  unsigned* Wfrot = Wfxz5 + (size_t)64 * KT7 * 256;     // 11 x 16
  unsigned* Wfkoo = Wfrot + (size_t)11 * KT5 * 256;     // 42 x 16
  // activation fragment buffers
  unsigned* Ax = Wfkoo + (size_t)42 * KT5 * 256;        // 256 x 23 tiles
  unsigned* Az = Ax + (size_t)(Mm / 16) * KT7 * 256;    // 256 x 16 tiles
  unsigned* Ah = Az + (size_t)(Mm / 16) * KT5 * 256;    // 256 x 16 tiles

  const dim3 blk(TPB);
  auto gpack = [](int nt, int kt) { return dim3((nt * kt * 32 + TPB - 1) / TPB); };
  const dim3 gax((Mm / 16) * KT7 * 32 / TPB);           // 736 blocks
  const dim3 gaz((Mm / 16) * KT5 * 32 / TPB);           // 512 blocks

  prep_x<<<(Mm * LlP + TPB - 1) / TPB, blk, 0, stream>>>(x_bsd, X, Mm * LlP);
  scale01<<<(Mm * Dd + TPB - 1) / TPB, blk, 0, stream>>>(z0, Z, Mm * Dd);
  scale01<<<(Mm * Pp + TPB - 1) / TPB, blk, 0, stream>>>(y0, Y, Mm * Pp);

  packW<<<gpack(32, KT7), blk, 0, stream>>>(W_h,     WfH,   Ll, 512,  KT7, 32);
  packW<<<gpack(64, KT5), blk, 0, stream>>>(W_ssz,   Wfssz, Dd, 1024, KT5, 64);
  packW<<<gpack(90, KT5), blk, 0, stream>>>(W_zx_v0, Wfzx0, Dd, 1440, KT5, 90);
  packW<<<gpack(64, KT7), blk, 0, stream>>>(W_xz_v2, Wfxz2, Ll, 1024, KT7, 64);
  packW<<<gpack(90, KT5), blk, 0, stream>>>(W_zx_v2, Wfzx2, Dd, 1440, KT5, 90);
  packW<<<gpack(64, KT7), blk, 0, stream>>>(W_xz_v3, Wfxz3, Ll, 1024, KT7, 64);
  packW<<<gpack(90, KT5), blk, 0, stream>>>(W_zx_v3, Wfzx3, Dd, 1440, KT5, 90);
  packW<<<gpack(64, KT7), blk, 0, stream>>>(W_xz_v4, Wfxz4, Ll, 1024, KT7, 64);
  packW<<<gpack(42, KT5), blk, 0, stream>>>(W_zy_v4, Wfzy4, Dd, 672,  KT5, 42);
  packW<<<gpack(64, KT7), blk, 0, stream>>>(W_xz_v5, Wfxz5, Ll, 1024, KT7, 64);
  packW<<<gpack(11, KT5), blk, 0, stream>>>(W_rot,   Wfrot, Dd, 168,  KT5, 11);
  packW<<<gpack(42, KT5), blk, 0, stream>>>(W_koo,   Wfkoo, Dd, 672,  KT5, 42);

  const int gm = Mm / BM;                 // 16 m-blocks

  packA<LlP><<<gax, blk, 0, stream>>>(X, Ax);
  gemm_gelu<LlP, Hh><<<dim3(gm, 32), blk, 0, stream>>>(Ax, WfH, H);
  packA<Dd><<<gaz, blk, 0, stream>>>(H, Ah);

  affine_gemm<Dd,  Dd, Dd ><<<dim3(gm, 32), blk, 0, stream>>>(Ah, Wfssz, Z, Z);
  packA<Dd><<<gaz, blk, 0, stream>>>(Z, Az);
  affine_gemm<Dd,  Ll, LlP><<<dim3(gm, 45), blk, 0, stream>>>(Az, Wfzx0, X, X);
  packA<LlP><<<gax, blk, 0, stream>>>(X, Ax);
  affine_gemm<LlP, Dd, Dd ><<<dim3(gm, 32), blk, 0, stream>>>(Ax, Wfxz2, Z, Z);
  packA<Dd><<<gaz, blk, 0, stream>>>(Z, Az);
  affine_gemm<Dd,  Ll, LlP><<<dim3(gm, 45), blk, 0, stream>>>(Az, Wfzx2, X, X);
  packA<LlP><<<gax, blk, 0, stream>>>(X, Ax);
  affine_gemm<LlP, Dd, Dd ><<<dim3(gm, 32), blk, 0, stream>>>(Ax, Wfxz3, Z, Z);
  packA<Dd><<<gaz, blk, 0, stream>>>(Z, Az);
  affine_gemm<Dd,  Ll, LlP><<<dim3(gm, 45), blk, 0, stream>>>(Az, Wfzx3, X, X);
  packA<LlP><<<gax, blk, 0, stream>>>(X, Ax);
  affine_gemm<LlP, Dd, Dd ><<<dim3(gm, 32), blk, 0, stream>>>(Ax, Wfxz4, Z, Z);
  packA<Dd><<<gaz, blk, 0, stream>>>(Z, Az);
  affine_gemm<Dd,  Pp, Pp ><<<dim3(gm, 21), blk, 0, stream>>>(Az, Wfzy4, Y, Y);
  affine_gemm<LlP, Dd, Dd ><<<dim3(gm, 32), blk, 0, stream>>>(Ax, Wfxz5, Z, Z);
  packA<Dd><<<gaz, blk, 0, stream>>>(Z, Az);

  gemm_plain<Dd, P2 ><<<dim3(gm, 11), blk, 0, stream>>>(Az, Wfrot, R);
  gemm_plain<Dd, 672><<<dim3(gm, 42), blk, 0, stream>>>(Az, Wfkoo, G);

  final_mix<<<(Mm * P2 + TPB - 1) / TPB, blk, 0, stream>>>(Y, R, G, a_y, b_y,
                                                           (float*)d_out, Mm * P2);
}
